// GAT_31387620999719
// MI455X (gfx1250) — matrix operand order repaired and checked by reading, not compiled, not run on hardware
//
#include <hip/hip_runtime.h>
#include <hip/hip_bf16.h>

typedef __attribute__((ext_vector_type(16))) _Float16 v16h;
typedef __attribute__((ext_vector_type(8)))  _Float16 v8h;
typedef __attribute__((ext_vector_type(8)))  float    v8f;

#define BB  4
#define NN  2048
#define DD  256
#define HH  8
#define DHH 32

// ---------- fragment helpers (wave32, 16x16x32 f16 WMMA) ----------

// A fragment: rows r0..r0+15, K-chunk kc..kc+31 from row-major f16 matrix (ld elems).
// Lane layout: lane&15 = row; lanes 0-15 hold K 0..15 contiguous, lanes 16-31 hold K 16..31.
__device__ __forceinline__ v16h load_a(const _Float16* __restrict__ p0, int r0, int kc, int ld, int lane) {
    const _Float16* p = p0 + (size_t)(r0 + (lane & 15)) * ld + kc + ((lane >> 4) << 4);
    v8h lo = *(const v8h*)p;          // K + {0..7}   (hi lanes: +16..23)
    v8h hi = *(const v8h*)(p + 8);    // K + {8..15}  (hi lanes: +24..31)
    v16h a;
#pragma unroll
    for (int i = 0; i < 8; ++i) { a[i] = lo[i]; a[i + 8] = hi[i]; }
    return a;
}

// B fragment from B^T stored row-major (ld elems): Bt[n][k]. Lane&15 = column n;
// lanes 0-15 hold K 0..15 contiguous, lanes 16-31 hold K 16..31 (same k order as load_a).
__device__ __forceinline__ v16h load_bt(const _Float16* __restrict__ p0, int n0, int kc, int ld, int lane) {
    const _Float16* p = p0 + (size_t)(n0 + (lane & 15)) * ld + kc + ((lane >> 4) << 4);
    v8h lo = *(const v8h*)p;
    v8h hi = *(const v8h*)(p + 8);
    v16h b;
#pragma unroll
    for (int i = 0; i < 8; ++i) { b[i] = lo[i]; b[i + 8] = hi[i]; }
    return b;
}

__device__ __forceinline__ v8f wmma16(v16h a, v16h b, v8f c) {
    return __builtin_amdgcn_wmma_f32_16x16x32_f16(false, a, false, b, (short)0, c, false, false);
}

// XOR-16 half-wave exchange via ds_swizzle (group-of-32: and=31, or=0, xor=16)
__device__ __forceinline__ float xor16(float v) {
    return __int_as_float(__builtin_amdgcn_ds_swizzle(__float_as_int(v), 0x401F));
}

// CDNA5 async global->LDS copy, 16 bytes per lane (ASYNCcnt path)
__device__ __forceinline__ void async_copy16(unsigned lds_off, const _Float16* g) {
    asm volatile("global_load_async_to_lds_b128 %0, %1, off"
                 :: "v"(lds_off), "v"((unsigned long long)(uintptr_t)g)
                 : "memory");
}
__device__ __forceinline__ void wait_async0() {
    asm volatile("s_wait_asynccnt 0" ::: "memory");
}

// ---------- kernel 1: convert x -> f16, weights -> f16 transposed ----------
__global__ void prep_kernel(const float* __restrict__ x,
                            const float* __restrict__ Wq, const float* __restrict__ Wk,
                            const float* __restrict__ Wv, const float* __restrict__ Wo,
                            const float* __restrict__ Wc,
                            _Float16* __restrict__ xh, _Float16* __restrict__ wt) {
    int idx = blockIdx.x * 256 + threadIdx.x;
    if (idx < BB * NN * DD) xh[idx] = (_Float16)x[idx];
    if (idx < 5 * DD * DD) {
        int w = idx >> 16;
        int r = idx & 65535;
        int i = r >> 8, j = r & 255;
        const float* W = (w == 0) ? Wq : (w == 1) ? Wk : (w == 2) ? Wv : (w == 3) ? Wo : Wc;
        wt[(w << 16) + (j << 8) + i] = (_Float16)W[(i << 8) + j];
    }
}

// ---------- kernel 2: fused Q/K/V projection ----------
__global__ void qkv_kernel(const _Float16* __restrict__ xh, const _Float16* __restrict__ wt,
                           const float* __restrict__ bq, const float* __restrict__ bk,
                           const float* __restrict__ bv,
                           _Float16* __restrict__ Qb, _Float16* __restrict__ Kb,
                           _Float16* __restrict__ Vt) {
    int lane = threadIdx.x & 31;
    int wave = threadIdx.x >> 5;
    int tile = blockIdx.x * 8 + wave;       // 3 * 512 * 16 = 24576 tiles
    int rt  = tile & 511;
    int ct  = (tile >> 9) & 15;
    int mat = tile >> 13;                    // 0=Q 1=K 2=V
    const _Float16* W = wt + (mat << 16);
    const float* bias = (mat == 0) ? bq : (mat == 1) ? bk : bv;
    int t0 = rt << 4, f0 = ct << 4;

    v8f c = {};
#pragma unroll
    for (int kc = 0; kc < DD; kc += 32) {
        v16h a = load_a(xh, t0, kc, DD, lane);
        v16h b = load_bt(W, f0, kc, DD, lane);
        c = wmma16(a, b, c);
    }
    int n  = lane & 15;
    int f  = f0 + n;
    float bval = bias[f];
    int h  = f >> 5, dh = f & 31;
    int mbase = (lane >> 4) << 3;
    // 1/sqrt(32) * log2(e): scores become base-2 exponents (v_exp_f32 is 2^x)
    const float scale = 0.17677669529663687f * 1.4426950408889634f;
#pragma unroll
    for (int i = 0; i < 8; ++i) {
        int t = t0 + mbase + i;
        int bidx = t >> 11, ntok = t & 2047;
        float v = c[i] + bval;
        int base = bidx * HH + h;
        if (mat == 0)      Qb[((size_t)base * NN + ntok) * DHH + dh] = (_Float16)(v * scale);
        else if (mat == 1) Kb[((size_t)base * NN + ntok) * DHH + dh] = (_Float16)v;
        else               Vt[((size_t)base * DHH + dh) * NN + ntok] = (_Float16)v;
    }
}

// ---------- kernel 3: flash attention, async-LDS staged K/V shared by 8 waves ----------
__global__ void attn_kernel(const _Float16* __restrict__ Qb, const _Float16* __restrict__ Kb,
                            const _Float16* __restrict__ Vt, _Float16* __restrict__ ctxh) {
    __shared__ _Float16 sK[2][32 * 32];   // 32 keys x 32 dh (row-major, contiguous in Kb)
    __shared__ _Float16 sV[2][32 * 32];   // 32 dh x 32 keys (rows of Vt)

    int lane = threadIdx.x & 31;
    int wave = threadIdx.x >> 5;
    // block handles 8 consecutive qtiles of one (b,h): 512 blocks total
    int lin = blockIdx.x;
    int qg = lin & 15;
    int h  = (lin >> 4) & 7;
    int b  = lin >> 7;
    const _Float16* qp = Qb + (size_t)(b * HH + h) * NN * DHH;
    const _Float16* kp = Kb + (size_t)(b * HH + h) * NN * DHH;
    const _Float16* vp = Vt + (size_t)(b * HH + h) * DHH * NN;
    int q0 = ((qg << 3) | wave) << 4;

    // B operand of S^T = K * Q^T : lane n holds Q row n (dh contiguous)
    v16h qf = load_bt(qp, q0, 0, DHH, lane);

    float m = -3.0e38f, l = 0.0f;
    v8f acc0 = {}, acc1 = {};                // ctx^T: dh 0-15 and dh 16-31 (rows) x 16 queries (cols)
    bool isLo = lane < 16;
    int tid = threadIdx.x;

    // cooperative async stage of one 32-key chunk (K: 2KB contiguous, V: 32 strided rows)
    auto stage = [&](int buf, int j) {
        if (tid < 128) {                                  // waves 0-3: K tile
            async_copy16((unsigned)(uintptr_t)&sK[buf][tid << 3],
                         kp + (size_t)j * DHH + (tid << 3));
        } else {                                          // waves 4-7: V tile
            int t = tid - 128;
            int row = t >> 2, c = (t & 3) << 3;
            async_copy16((unsigned)(uintptr_t)&sV[buf][(row << 5) + c],
                         vp + (size_t)row * NN + j + c);
        }
    };

    stage(0, 0);
    for (int jc = 0; jc < NN / 32; ++jc) {
        int cur = jc & 1;
        wait_async0();        // this wave's staged chunk is in LDS
        __syncthreads();      // all waves' chunks in LDS; prev compute done -> other buf reusable
        if (jc + 1 < NN / 32) stage(cur ^ 1, (jc + 1) << 5);

        const _Float16* kb = &sK[cur][0];
        const _Float16* vb = &sV[cur][0];
        v16h ka0 = load_a(kb, 0,  0, 32, lane);           // keys 0-15 of chunk
        v16h ka1 = load_a(kb, 16, 0, 32, lane);           // keys 16-31
        v8f z = {};
        v8f s0 = wmma16(ka0, qf, z);                      // S^T tile (keys x queries), base-2 scaled
        v8f s1 = wmma16(ka1, qf, z);

        // online softmax over keys (per query column = per lane&15), base 2
        float mx = s0[0];
#pragma unroll
        for (int i = 1; i < 8; ++i) mx = fmaxf(mx, s0[i]);
#pragma unroll
        for (int i = 0; i < 8; ++i) mx = fmaxf(mx, s1[i]);
        mx = fmaxf(mx, xor16(mx));                        // combine key-halves
        float mnew = fmaxf(m, mx);
        float corr = exp2f(m - mnew);
        m = mnew;
        float ps = 0.0f;
#pragma unroll
        for (int i = 0; i < 8; ++i) {
            s0[i] = exp2f(s0[i] - mnew); ps += s0[i];
            s1[i] = exp2f(s1[i] - mnew); ps += s1[i];
        }
        l = l * corr + ps;
#pragma unroll
        for (int i = 0; i < 8; ++i) { acc0[i] *= corr; acc1[i] *= corr; }

        // Repack exp(S^T) C-fragments into the B-fragment of P^T (32 keys x 16 queries).
        // lo lane needs K0-15: own s0 (K0-7) + partner-hi s0 (K8-15)
        // hi lane needs K16-31: partner-lo s1 (K16-23) + own s1 (K24-31)
        v16h pf;
#pragma unroll
        for (int i = 0; i < 8; ++i) {
            float xi = isLo ? s0[i] : s1[i];
            float zi = isLo ? s1[i] : s0[i];
            float yi = xor16(zi);
            pf[i]     = (_Float16)(isLo ? xi : yi);
            pf[i + 8] = (_Float16)(isLo ? yi : xi);
        }

        // ctx^T += V^T(dh x keys) * P^T(keys x queries)
        v16h va0 = load_a(vb, 0,  0, 32, lane);
        v16h va1 = load_a(vb, 16, 0, 32, lane);
        acc0 = wmma16(va0, pf, acc0);
        acc1 = wmma16(va1, pf, acc1);
    }

    l += xor16(l);
    float inv = 1.0f / l;
    int qcol  = lane & 15;
    int mbase = (lane >> 4) << 3;
    size_t rowbase = ((size_t)b * NN + q0 + qcol) * DD + h * DHH;
#pragma unroll
    for (int i = 0; i < 8; ++i) {
        ctxh[rowbase + mbase + i]      = (_Float16)(acc0[i] * inv);
        ctxh[rowbase + 16 + mbase + i] = (_Float16)(acc1[i] * inv);
    }
}

// ---------- kernel 4: t = ctx @ Wo + bo (f16 out) ----------
__global__ void proj1_kernel(const _Float16* __restrict__ ctxh, const _Float16* __restrict__ wt,
                             const float* __restrict__ bo, _Float16* __restrict__ th) {
    int lane = threadIdx.x & 31;
    int wave = threadIdx.x >> 5;
    int tile = blockIdx.x * 8 + wave;   // 512*16 = 8192 tiles
    int rt = tile & 511;
    int ct = tile >> 9;
    const _Float16* W = wt + (3 << 16);  // WoT
    int t0 = rt << 4, f0 = ct << 4;
    v8f c = {};
#pragma unroll
    for (int kc = 0; kc < DD; kc += 32) {
        v16h a = load_a(ctxh, t0, kc, DD, lane);
        v16h b = load_bt(W, f0, kc, DD, lane);
        c = wmma16(a, b, c);
    }
    int n = lane & 15;
    float bval = bo[f0 + n];
    int mbase = (lane >> 4) << 3;
#pragma unroll
    for (int i = 0; i < 8; ++i) {
        int t = t0 + mbase + i;
        th[(size_t)t * DD + f0 + n] = (_Float16)(c[i] + bval);
    }
}

// ---------- kernel 5: out = x + t @ Wc + bc (f32 out) ----------
__global__ void proj2_kernel(const _Float16* __restrict__ th, const _Float16* __restrict__ wt,
                             const float* __restrict__ bc, const float* __restrict__ x,
                             float* __restrict__ out) {
    int lane = threadIdx.x & 31;
    int wave = threadIdx.x >> 5;
    int tile = blockIdx.x * 8 + wave;
    int rt = tile & 511;
    int ct = tile >> 9;
    const _Float16* W = wt + (4 << 16);  // WcT
    int t0 = rt << 4, f0 = ct << 4;
    v8f c = {};
#pragma unroll
    for (int kc = 0; kc < DD; kc += 32) {
        v16h a = load_a(th, t0, kc, DD, lane);
        v16h b = load_bt(W, f0, kc, DD, lane);
        c = wmma16(a, b, c);
    }
    int n = lane & 15;
    float bval = bc[f0 + n];
    int mbase = (lane >> 4) << 3;
#pragma unroll
    for (int i = 0; i < 8; ++i) {
        size_t idx = (size_t)(t0 + mbase + i) * DD + f0 + n;
        out[idx] = x[idx] + c[i] + bval;
    }
}

extern "C" void kernel_launch(void* const* d_in, const int* in_sizes, int n_in,
                              void* d_out, int out_size, void* d_ws, size_t ws_size,
                              hipStream_t stream) {
    const float* x  = (const float*)d_in[0];
    const float* Wq = (const float*)d_in[1];
    const float* bq = (const float*)d_in[2];
    const float* Wk = (const float*)d_in[3];
    const float* bk = (const float*)d_in[4];
    const float* Wv = (const float*)d_in[5];
    const float* bv = (const float*)d_in[6];
    const float* Wo = (const float*)d_in[7];
    const float* bo = (const float*)d_in[8];
    const float* Wc = (const float*)d_in[9];
    const float* bc = (const float*)d_in[10];
    float* out = (float*)d_out;

    char* ws = (char*)d_ws;
    _Float16* wt   = (_Float16*)(ws + 0);         //  5*65536 f16  = 640 KB
    _Float16* xh   = (_Float16*)(ws + 655360);    //  8192*256 f16 = 4 MB
    _Float16* Qb   = (_Float16*)(ws + 4849664);   //  4 MB
    _Float16* Kb   = (_Float16*)(ws + 9043968);   //  4 MB
    _Float16* Vt   = (_Float16*)(ws + 13238272);  //  4 MB
    _Float16* ctxh = (_Float16*)(ws + 17432576);  //  4 MB
    _Float16* th   = (_Float16*)(ws + 21626880);  //  4 MB  (total ~24.6 MB)

    prep_kernel <<<8192, 256, 0, stream>>>(x, Wq, Wk, Wv, Wo, Wc, xh, wt);
    qkv_kernel  <<<3072, 256, 0, stream>>>(xh, wt, bq, bk, bv, Qb, Kb, Vt);
    attn_kernel <<<512,  256, 0, stream>>>(Qb, Kb, Vt, ctxh);
    proj1_kernel<<<1024, 256, 0, stream>>>(ctxh, wt, bo, th);
    proj2_kernel<<<1024, 256, 0, stream>>>(th, wt, bc, x, out);
}
